// BERTgridGenerator_39831526703157
// MI455X (gfx1250) — matrix operand order repaired and checked by reading, not compiled, not run on hardware
//
#include <hip/hip_runtime.h>
#include <hip/hip_bf16.h>

typedef __attribute__((ext_vector_type(16))) _Float16 v16h;
typedef __attribute__((ext_vector_type(8)))  float    v8f;

// Problem constants (match reference module constants)
constexpr int Bn    = 8;
constexpr int Sn    = 510;
constexpr int Dn    = 768;
constexpr int Kn    = 170;
constexpr int KPAD  = 176;   // 11 * 16
constexpr int KT    = 11;    // k-tiles of 16
constexpr int ST    = 512;   // S padded to multiple of 32
constexpr int STRD  = 8;
constexpr int Hc    = 1024 / STRD;  // 128
constexpr int Wc    = 768  / STRD;  // 96

// gfx1250 async global->LDS staging (ASYNCcnt path), guarded so compilation
// cannot regress on toolchains that lack the builtins.  The builtin expects
// typed address-space pointers (diagnostic showed param type '__device__ int*').
#if __has_builtin(__builtin_amdgcn_global_load_async_to_lds_b32) && \
    __has_builtin(__builtin_amdgcn_s_wait_asynccnt)
#define USE_ASYNC_LDS 1
typedef __attribute__((address_space(1))) int gint;   // global (device) int
typedef __attribute__((address_space(3))) int lint;   // LDS (shared) int
#endif

// ---------------------------------------------------------------------------
// Kernel 1: segsum[b,k,d] = sum_s onehot(seg[b,s]==k) * tok[b,s,d]
// via V_WMMA_F32_16X16X32_F16.  One wave per (dtile, b); 11 accumulators
// cover all K rows, B-fragment (tok) built once per 32-token step and reused
// across all 11 WMMAs.  A fragments double-buffered to avoid the WMMA->VALU
// WAR hazard NOPs (ISA 7.12.1: 4 co-exec slots for *F16 WMMA).
// ---------------------------------------------------------------------------
__global__ __launch_bounds__(32) void segsum_wmma_kernel(
    const float* __restrict__ emb,     // [VOCAB, D]
    const int*   __restrict__ corpus,  // [B, S]
    const int*   __restrict__ mask,    // [B, S]
    const int*   __restrict__ seg,     // [B, S]
    float*       __restrict__ segsum)  // [B, KPAD, D]  (workspace)
{
    const int dtile = blockIdx.x;          // 0..47
    const int b     = blockIdx.y;          // 0..7
    const int lane  = threadIdx.x;         // 0..31
    const bool lo   = lane < 16;
    const int  nl   = lane & 15;
    const int  dcol = dtile * 16 + nl;

    __shared__ int s_seg[32];
    __shared__ int s_tok[32];
    __shared__ int s_msk[32];

    v8f acc[KT];
    #pragma unroll
    for (int t = 0; t < KT; ++t)
        #pragma unroll
        for (int r = 0; r < 8; ++r) acc[t][r] = 0.0f;

    for (int s0 = 0; s0 < ST; s0 += 32) {
        const int s = s0 + lane;
        const bool ok = (s < Sn);
#ifdef USE_ASYNC_LDS
        // Async global->LDS (no VGPR round trip); clamp the tail addresses and
        // patch sentinels after the wait so padding semantics stay exact.
        const int sc = ok ? s : (Sn - 1);
        __builtin_amdgcn_global_load_async_to_lds_b32(
            (gint*)(seg + b * Sn + sc),    (lint*)&s_seg[lane], 0, 0);
        __builtin_amdgcn_global_load_async_to_lds_b32(
            (gint*)(corpus + b * Sn + sc), (lint*)&s_tok[lane], 0, 0);
        __builtin_amdgcn_global_load_async_to_lds_b32(
            (gint*)(mask + b * Sn + sc),   (lint*)&s_msk[lane], 0, 0);
        __builtin_amdgcn_s_wait_asynccnt(0);
        if (!ok) { s_seg[lane] = -1; s_msk[lane] = 0; }
#else
        s_seg[lane] = ok ? seg[b * Sn + s]    : -1;
        s_tok[lane] = ok ? corpus[b * Sn + s] : 0;
        s_msk[lane] = ok ? mask[b * Sn + s]   : 0;
#endif
        __syncthreads();

        // B fragment: tok = emb_table[corpus] * mask, f16.
        // Layout used here (contiguous k, same order as A): VGPR j, half h; lanes0-15 hold
        // K=0..15 (kk=2j+h), lanes16-31 hold K=16..31.
        v16h bf;
        #pragma unroll
        for (int e = 0; e < 16; ++e) {
            const int j = e >> 1, h = e & 1;
            const int kk = (lo ? 0 : 16) + 2 * j + h;
            const int c  = s_tok[kk];
            const float v = (float)s_msk[kk] * emb[(long)c * Dn + dcol];
            bf[e] = (_Float16)v;
        }

        // A fragments (onehot) per k-tile; double-buffered registers.
        // Layout used here (contiguous k, same order as B): lanes0-15 M=lane, K=0..15;
        // lanes16-31 M=lane-16, K=16..31.
        v16h af0, af1;
        #pragma unroll
        for (int t = 0; t < KT; ++t) {
            v16h& af = (t & 1) ? af1 : af0;
            const int kbase = t * 16;
            #pragma unroll
            for (int e = 0; e < 16; ++e) {
                const int j = e >> 1, h = e & 1;
                const int base = 2 * j + h;
                const int kk = base + (lo ? 0 : 16);
                af[e] = (_Float16)((s_seg[kk] == kbase + nl) ? 1.0f : 0.0f);
            }
            acc[t] = __builtin_amdgcn_wmma_f32_16x16x32_f16(
                false, af, false, bf, (short)0, acc[t], false, false);
        }
        __syncthreads();
    }

    // C layout (32-bit 16x16): VGPR r -> M = r (+8 for upper lanes), N = lane&15
    #pragma unroll
    for (int t = 0; t < KT; ++t) {
        #pragma unroll
        for (int r = 0; r < 8; ++r) {
            const int row = t * 16 + r + (lo ? 0 : 8);
            segsum[((long)b * KPAD + row) * Dn + dcol] = acc[t][r];
        }
    }
}

// ---------------------------------------------------------------------------
// Kernel 2: agg[b,0,:] = 0 ; agg[b,k+1,:] = (segsum[b,k,:] - tok[first_k])/count_k
// ---------------------------------------------------------------------------
__global__ __launch_bounds__(256) void agg_finalize_kernel(
    const float* __restrict__ emb,
    const int*   __restrict__ corpus,
    const int*   __restrict__ mask,
    const int*   __restrict__ seg,
    const float* __restrict__ segsum,  // [B, KPAD, D]
    float*       __restrict__ agg)     // [B, K, D]
{
    const int kout = blockIdx.x % Kn;
    const int b    = blockIdx.x / Kn;
    float* out = agg + ((long)b * Kn + kout) * Dn;

    if (kout == 0) {
        for (int d = threadIdx.x; d < Dn; d += blockDim.x) out[d] = 0.0f;
        return;
    }
    const int k = kout - 1;

    __shared__ int s_cnt, s_first;
    if (threadIdx.x == 0) {
        int cnt = 0, first = -1;
        for (int s = 0; s < Sn; ++s) {
            if (seg[b * Sn + s] == k) { if (first < 0) first = s; ++cnt; }
        }
        s_cnt = cnt; s_first = first;
    }
    __syncthreads();

    const int cnt = s_cnt, first = s_first;
    const int c   = (first >= 0) ? corpus[b * Sn + first] : 0;
    const float m = (first >= 0) ? (float)mask[b * Sn + first] : 0.0f;
    const float inv = (cnt > 0) ? 1.0f / (float)cnt : 0.0f;

    for (int d = threadIdx.x; d < Dn; d += blockDim.x) {
        const float fe = (first >= 0) ? emb[(long)c * Dn + d] * m : 0.0f;
        out[d] = (segsum[((long)b * KPAD + k) * Dn + d] - fe) * inv;
    }
}

// ---------------------------------------------------------------------------
// Kernel 3: rasterize boxes (last-write-wins == max seg id), then stream the
// [B, D, Hc, Wc] output.  x-contiguous coalesced non-temporal stores (302 MB
// output > 192 MB L2 -> bypass); agg gathers stay cached (4 MB working set).
// This kernel is the roofline term: ~302 MB stores -> ~13 us at 23.3 TB/s.
// ---------------------------------------------------------------------------
__global__ __launch_bounds__(Wc) void grid_paint_kernel(
    const int*   __restrict__ coor,   // [B, K, 4]
    const float* __restrict__ agg,    // [B, K, D]
    float*       __restrict__ grid)   // [B, D, Hc, Wc]
{
    const int y = blockIdx.x % Hc;
    const int b = blockIdx.x / Hc;
    const int x = threadIdx.x;        // 0..95

    __shared__ int4 s_box[Kn];
    for (int k = threadIdx.x; k < Kn; k += blockDim.x)
        s_box[k] = ((const int4*)coor)[b * Kn + k];
    __syncthreads();

    int segid = -1;
    for (int k = 0; k < Kn; ++k) {
        const int4 cc = s_box[k];
        const int x1 = cc.x / STRD, y1 = cc.y / STRD;
        const int x2 = cc.z / STRD, y2 = cc.w / STRD;
        const bool in = (y >= y1) & (y < y2) & (x >= x1) & (x < x2);
        if (in) segid = k;            // ascending k => last write wins
    }

    const float* arow = agg + ((long)b * Kn + (segid >= 0 ? segid : 0)) * Dn;
    const bool hit = (segid >= 0);
    if (hit) __builtin_prefetch(arow, 0, 1);

    float* o = grid + ((long)b * Dn) * (Hc * Wc) + (long)y * Wc + x;
    for (int d = 0; d < Dn; ++d) {
        const float v = hit ? arow[d] : 0.0f;
        __builtin_nontemporal_store(v, o);
        o += (long)Hc * Wc;
    }
}

// ---------------------------------------------------------------------------
extern "C" void kernel_launch(void* const* d_in, const int* in_sizes, int n_in,
                              void* d_out, int out_size, void* d_ws, size_t ws_size,
                              hipStream_t stream) {
    const float* emb    = (const float*)d_in[0];
    const int*   corpus = (const int*)d_in[1];
    const int*   mask   = (const int*)d_in[2];
    const int*   seg    = (const int*)d_in[3];
    const int*   coor   = (const int*)d_in[4];
    (void)in_sizes; (void)n_in; (void)out_size; (void)ws_size;

    float* agg    = (float*)d_out;                          // [B,K,D]
    float* grid   = (float*)d_out + (size_t)Bn * Kn * Dn;   // [B,D,Hc,Wc]
    float* segsum = (float*)d_ws;                           // [B,KPAD,D] = 4.3 MB

    segsum_wmma_kernel<<<dim3(Dn / 16, Bn), 32, 0, stream>>>(emb, corpus, mask, seg, segsum);
    agg_finalize_kernel<<<Bn * Kn, 256, 0, stream>>>(emb, corpus, mask, seg, segsum, agg);
    grid_paint_kernel<<<Bn * Hc, Wc, 0, stream>>>(coor, agg, grid);
}
